// AttentionHead_30047591203249
// MI455X (gfx1250) — matrix operand order repaired and checked by reading, not compiled, not run on hardware
//
#include <hip/hip_runtime.h>

// ---------- types ----------
typedef __attribute__((ext_vector_type(16))) __bf16 bf16x16;
typedef __attribute__((ext_vector_type(8)))  float  f32x8;
typedef __attribute__((ext_vector_type(4)))  int    v4i_t;

union BF16V {
    bf16x16 v;
    uint4   u[2];
};

#define WMMA_BF16(A, B, C) \
    __builtin_amdgcn_wmma_f32_16x16x32_bf16(false, (A), false, (B), (short)0, (C), false, false)

// ---------- CDNA5 async global->LDS path (guarded; falls back to sync copy) ----------
#if defined(__AMDGCN__) && defined(__gfx1250__)
  #if __has_builtin(__builtin_amdgcn_global_load_async_to_lds_b128)
    #define USE_ASYNC_LDS 1
  #endif
#endif

#ifdef USE_ASYNC_LDS
  #define ASYNC_B128(SRC, DST)                                              \
      __builtin_amdgcn_global_load_async_to_lds_b128(                       \
          (__attribute__((address_space(1))) v4i_t*)(SRC),                  \
          (__attribute__((address_space(3))) v4i_t*)(DST), 0, 0)
  #if __has_builtin(__builtin_amdgcn_s_wait_asynccnt)
    #define ASYNC_WAIT(N) __builtin_amdgcn_s_wait_asynccnt((unsigned short)(N))
  #else
    #define ASYNC_WAIT(N) asm volatile("s_wait_asynccnt %0" :: "n"(N) : "memory")
  #endif
#else
  #define ASYNC_WAIT(N) ((void)0)
#endif

__device__ __forceinline__ void cvt8(bf16x16& d, int base, float4 a, float4 b) {
    d[base + 0] = (__bf16)a.x; d[base + 1] = (__bf16)a.y;
    d[base + 2] = (__bf16)a.z; d[base + 3] = (__bf16)a.w;
    d[base + 4] = (__bf16)b.x; d[base + 5] = (__bf16)b.y;
    d[base + 6] = (__bf16)b.z; d[base + 7] = (__bf16)b.w;
}

// Problem constants
#define BATCH 4
#define SEQ   4096
#define DMODEL 2048
#define DATT  128
#define MROWS (BATCH * SEQ)      // 16384
#define NCOLS (3 * DATT)         // 384
#define NTILES (NCOLS / 16)      // 24

// ==========================================================================
// Kernel 1: qkv = x @ W^T  (M=16384, N=384, K=2048), fp32 -> bf16 outputs.
// One 16x16 tile per wave via v_wmma_f32_16x16x32_bf16.
// Q gets the 1/sqrt(128) softmax scale folded in; V is stored transposed
// ([b][d][s]) so the attention kernel's P*V B-operands are contiguous.
// ==========================================================================
__global__ __launch_bounds__(256) void qkv_proj_kernel(
    const float* __restrict__ x, const float* __restrict__ wq,
    __bf16* __restrict__ Qb, __bf16* __restrict__ Kb, __bf16* __restrict__ Vtb)
{
    const int lane = threadIdx.x & 31;
    const int wave = threadIdx.x >> 5;
    const int tile = blockIdx.x * 8 + wave;          // 24576 tiles total
    const int mt = tile / NTILES;
    const int nt = tile - mt * NTILES;
    const int mrow0 = mt * 16;
    const int ncol0 = nt * 16;
    const int col  = lane & 15;
    const int half = lane >> 4;
    const int sub8 = half * 8;

    // A (x): lane L<16 -> row L, K in {0..7,16..23}; lane>=16 -> K in {8..15,24..31}
    const float* xr = x  + (size_t)(mrow0 + col) * DMODEL;
    // B (W): lane n<16 -> out-col n, K in {0..7,16..23}; lanes 16..31 -> K in {8..15,24..31}
    const float* wr = wq + (size_t)(ncol0 + col) * DMODEL + half * 8;

    f32x8 acc = {};
    for (int kc = 0; kc < DMODEL; kc += 32) {
        if (kc + 64 < DMODEL) {
            __builtin_prefetch(xr + kc + 64, 0, 3);
            __builtin_prefetch(wr + kc + 64, 0, 3);
        }
        BF16V a, bm;
        float4 f0 = *(const float4*)(xr + kc + sub8);
        float4 f1 = *(const float4*)(xr + kc + sub8 + 4);
        float4 f2 = *(const float4*)(xr + kc + 16 + sub8);
        float4 f3 = *(const float4*)(xr + kc + 16 + sub8 + 4);
        cvt8(a.v, 0, f0, f1);
        cvt8(a.v, 8, f2, f3);
        float4 g0 = *(const float4*)(wr + kc);
        float4 g1 = *(const float4*)(wr + kc + 4);
        float4 g2 = *(const float4*)(wr + kc + 16);
        float4 g3 = *(const float4*)(wr + kc + 20);
        cvt8(bm.v, 0, g0, g1);
        cvt8(bm.v, 8, g2, g3);
        acc = WMMA_BF16(a.v, bm.v, acc);
    }

    const float scale = 0.088388347648318447f;   // 1/sqrt(128)
    // C layout: VGPR r, lane n -> row = r + 8*(n>=16), col = n%16
    #pragma unroll
    for (int r = 0; r < 8; ++r) {
        int row = mrow0 + half * 8 + r;          // token index in [0, 16384)
        int n   = ncol0 + col;                   // output feature in [0, 384)
        float v = acc[r];
        if (n < DATT) {                          // Q (scaled)
            Qb[(size_t)row * DATT + n] = (__bf16)(v * scale);
        } else if (n < 2 * DATT) {               // K
            Kb[(size_t)row * DATT + (n - DATT)] = (__bf16)v;
        } else {                                 // V, transposed [b][d][s]
            int bi = row >> 12;                  // row / 4096
            int s  = row & 4095;
            Vtb[((size_t)bi * DATT + (n - 2 * DATT)) * SEQ + s] = (__bf16)v;
        }
    }
}

// ==========================================================================
// Tile staging: 32 keys x 128 dims of K (row-major) and 128 dims x 32 keys
// of V^T into LDS. 4 x b128 per thread -> async-to-LDS when available.
// ==========================================================================
__device__ __forceinline__ void stage_tiles(int tid,
    const __bf16* __restrict__ Kg, const __bf16* __restrict__ Vg,
    int kvbase, __bf16* Kd, __bf16* Vd)
{
    const uint4* ks = (const uint4*)(Kg + (size_t)kvbase * DATT);
    uint4* kd = (uint4*)Kd;
    uint4* vd = (uint4*)Vd;
    const int i0 = tid, i1 = tid + 256;
    const int dim0 = i0 >> 2, part0 = i0 & 3;
    const int dim1 = i1 >> 2, part1 = i1 & 3;
    const uint4* vs0 = (const uint4*)(Vg + (size_t)dim0 * SEQ + kvbase + part0 * 8);
    const uint4* vs1 = (const uint4*)(Vg + (size_t)dim1 * SEQ + kvbase + part1 * 8);
#ifdef USE_ASYNC_LDS
    ASYNC_B128(ks + i0, kd + i0);
    ASYNC_B128(ks + i1, kd + i1);
    ASYNC_B128(vs0, vd + i0);
    ASYNC_B128(vs1, vd + i1);
#else
    kd[i0] = ks[i0];
    kd[i1] = ks[i1];
    vd[i0] = *vs0;
    vd[i1] = *vs1;
#endif
}

// ==========================================================================
// Kernel 2: causal flash attention, double-buffered async K/V staging.
// Block = 256 threads (8 waves); each block owns 128 query rows of one batch,
// each wave a 16-row stripe. kv tiles of 32 keys.
// ==========================================================================
__global__ __launch_bounds__(256) void flash_attn_kernel(
    const __bf16* __restrict__ Qb, const __bf16* __restrict__ Kb,
    const __bf16* __restrict__ Vtb, float* __restrict__ out)
{
    __shared__ __align__(16) __bf16 Ksh[2][32 * DATT];   // [buf][key][dim]  16 KB
    __shared__ __align__(16) __bf16 Vsh[2][DATT * 32];   // [buf][dim][key]  16 KB
    __shared__ __align__(16) float  Psh[8 * 16 * 32];    // per-wave P       16 KB

    const int tid   = threadIdx.x;
    const int wv    = tid >> 5;
    const int lane  = tid & 31;
    const int b     = blockIdx.y;
    const int qb0   = blockIdx.x * 128;
    const int qrow0 = qb0 + wv * 16;        // this wave's first query row
    const int col   = lane & 15;
    const int half  = lane >> 4;
    const int rbase = half * 8;
    const int sub8  = half * 8;

    // ---- load Q A-operands (4 chunks of 16x32), scale already folded in ----
    BF16V aq[4];
    const __bf16* qrp = Qb + ((size_t)(b * SEQ + qrow0 + col)) * DATT;
    #pragma unroll
    for (int c = 0; c < 4; ++c) {
        aq[c].u[0] = *(const uint4*)(qrp + c * 32 + half * 16);
        aq[c].u[1] = *(const uint4*)(qrp + c * 32 + half * 16 + 8);
    }

    f32x8 oacc[8];
    f32x8 zero = {};
    #pragma unroll
    for (int dt = 0; dt < 8; ++dt) oacc[dt] = zero;

    float mrow[8], lrow[8];
    #pragma unroll
    for (int r = 0; r < 8; ++r) { mrow[r] = -1e30f; lrow[r] = 0.0f; }

    float* Pw = Psh + wv * (16 * 32);
    const __bf16* Kg = Kb  + (size_t)b * SEQ * DATT;
    const __bf16* Vg = Vtb + (size_t)b * DATT * SEQ;

    const int nsteps = (qb0 >> 5) + 4;   // kv tiles up to and incl. diagonal

    // prologue: stage tile 0 into buffer 0
    stage_tiles(tid, Kg, Vg, 0, Ksh[0], Vsh[0]);

    for (int j = 0; j < nsteps; ++j) {
        const int kvbase = j * 32;
        const int cur = j & 1;

        // prefetch next tile into the other buffer, then wait for tile j
        // (async loads complete in order; 4 ops per thread per tile)
        if (j + 1 < nsteps) {
            stage_tiles(tid, Kg, Vg, kvbase + 32, Ksh[cur ^ 1], Vsh[cur ^ 1]);
            ASYNC_WAIT(4);
        } else {
            ASYNC_WAIT(0);
        }
        __syncthreads();                 // tile j visible to all waves

        // wave-uniform skip of fully-masked kv tiles (EXEC stays all-1s)
        if (kvbase <= qrow0 + 15) {
            const __bf16* Kcur = Ksh[cur];
            const __bf16* Vcur = Vsh[cur];

            // ---- S = Q K^T : 16 queries x 32 keys, two 16x16 C tiles ----
            f32x8 s0 = {}, s1 = {};
            #pragma unroll
            for (int c = 0; c < 4; ++c) {
                BF16V b0, b1;
                const __bf16* kp = Kcur + (size_t)col * DATT + c * 32 + half * 16;
                b0.u[0] = *(const uint4*)(kp);
                b0.u[1] = *(const uint4*)(kp + 8);
                b1.u[0] = *(const uint4*)(kp + 16 * DATT);
                b1.u[1] = *(const uint4*)(kp + 16 * DATT + 8);
                s0 = WMMA_BF16(aq[c].v, b0.v, s0);
                s1 = WMMA_BF16(aq[c].v, b1.v, s1);
            }

            // ---- causal mask + online softmax (row stats across 16 lanes) ----
            #pragma unroll
            for (int r = 0; r < 8; ++r) {
                const int qrow = qrow0 + rbase + r;
                const int key0 = kvbase + col;
                if (key0 > qrow)      s0[r] = -1e30f;
                if (key0 + 16 > qrow) s1[r] = -1e30f;

                float mx = fmaxf(s0[r], s1[r]);
                mx = fmaxf(mx, __shfl_xor(mx, 1, 32));
                mx = fmaxf(mx, __shfl_xor(mx, 2, 32));
                mx = fmaxf(mx, __shfl_xor(mx, 4, 32));
                mx = fmaxf(mx, __shfl_xor(mx, 8, 32));
                float mnew  = fmaxf(mrow[r], mx);
                float alpha = __expf(mrow[r] - mnew);
                float p0 = __expf(s0[r] - mnew);
                float p1 = __expf(s1[r] - mnew);
                float rs = p0 + p1;
                rs += __shfl_xor(rs, 1, 32);
                rs += __shfl_xor(rs, 2, 32);
                rs += __shfl_xor(rs, 4, 32);
                rs += __shfl_xor(rs, 8, 32);
                lrow[r] = lrow[r] * alpha + rs;
                mrow[r] = mnew;
                #pragma unroll
                for (int dt = 0; dt < 8; ++dt) oacc[dt][r] *= alpha;
                // stash P (C-layout) into per-wave LDS scratch, row-major 16x32
                Pw[(rbase + r) * 32 + col]      = p0;
                Pw[(rbase + r) * 32 + 16 + col] = p1;
            }

            // ---- reload P in A-layout (16x32) as bf16 (same-wave LDS: in order) ----
            BF16V pa;
            const float* prow = Pw + col * 32;
            #pragma unroll
            for (int t = 0; t < 4; ++t) {
                float2 f = *(const float2*)(prow + half * 16 + 2 * t);
                float2 g = *(const float2*)(prow + half * 16 + 8 + 2 * t);
                pa.v[2 * t]     = (__bf16)f.x;
                pa.v[2 * t + 1] = (__bf16)f.y;
                pa.v[8 + 2 * t] = (__bf16)g.x;
                pa.v[9 + 2 * t] = (__bf16)g.y;
            }

            // ---- O += P V : 8 dim-tiles of 16 ----
            #pragma unroll
            for (int dt = 0; dt < 8; ++dt) {
                BF16V bvv;
                const __bf16* vp = Vcur + (size_t)(dt * 16 + col) * 32 + half * 16;
                bvv.u[0] = *(const uint4*)(vp);
                bvv.u[1] = *(const uint4*)(vp + 8);
                oacc[dt] = WMMA_BF16(pa.v, bvv.v, oacc[dt]);
            }
        }
        __syncthreads();                 // done reading buf[cur] before reuse
    }

    // ---- epilogue: O /= l, write fp32 output [b][s][128] ----
    float* outw = out + ((size_t)(b * SEQ + qrow0)) * DATT;
    #pragma unroll
    for (int r = 0; r < 8; ++r) {
        float inv = 1.0f / lrow[r];
        #pragma unroll
        for (int dt = 0; dt < 8; ++dt)
            outw[(rbase + r) * DATT + dt * 16 + col] = oacc[dt][r] * inv;
    }
}

// ==========================================================================
extern "C" void kernel_launch(void* const* d_in, const int* in_sizes, int n_in,
                              void* d_out, int out_size, void* d_ws, size_t ws_size,
                              hipStream_t stream) {
    (void)in_sizes; (void)n_in; (void)out_size; (void)ws_size;
    const float* x  = (const float*)d_in[0];   // [4,4096,2048] fp32
    const float* wq = (const float*)d_in[1];   // [384,2048] fp32

    // workspace: Q, K, V^T in bf16 (4 MB each = 12 MB total)
    __bf16* Qb  = (__bf16*)d_ws;
    __bf16* Kb  = Qb + (size_t)MROWS * DATT;
    __bf16* Vtb = Kb + (size_t)MROWS * DATT;
    float*  out = (float*)d_out;               // [4,4096,128] fp32

    // 24576 16x16 tiles, 8 waves per block -> 3072 blocks
    qkv_proj_kernel<<<3072, 256, 0, stream>>>(x, wq, Qb, Kb, Vtb);
    // 32 query blocks x 4 batches
    flash_attn_kernel<<<dim3(32, 4), 256, 0, stream>>>(Qb, Kb, Vtb, out);
}
